// IndividualAutoEncoder_71622874628203
// MI455X (gfx1250) — compile-verified, matrix operand order checked by reading, not run on hardware
//
#include <hip/hip_runtime.h>
#include <hip/hip_bf16.h>

#define Bdim 256
#define Sdim 512
#define Vdim 4
#define Hdim 512

typedef __attribute__((ext_vector_type(16))) __bf16 v16bf;
typedef __attribute__((ext_vector_type(8)))  float  v8f;

union TileU { v16bf v; uint4 q[2]; };

__device__ inline unsigned short f32_to_bf16(float f) {
    unsigned int u = __float_as_uint(f);
    unsigned int r = (u + 0x7FFFu + ((u >> 16) & 1u)) >> 16;
    return (unsigned short)r;
}

// Load one 16x32 bf16 tile operand in the CDNA5 WMMA A/B register layout.
// lane 0-15  : row (base+lane),    K = k0+0..7   and k0+16..23
// lane 16-31 : row (base+lane-16), K = k0+8..15  and k0+24..31
__device__ inline v16bf load_tile(const unsigned short* __restrict__ base,
                                  int row0, int k0, int lane) {
    int r = lane & 15;
    int half = lane >> 4;
    const unsigned short* p = base + (size_t)(row0 + r) * Hdim + k0 + half * 8;
    TileU t;
    t.q[0] = *(const uint4*)(p);
    t.q[1] = *(const uint4*)(p + 16);
    return t.v;
}

// ---------------- utility kernels ----------------
__global__ __launch_bounds__(256) void k_f32_to_bf16(const float* __restrict__ src,
                                                     unsigned short* __restrict__ dst, int n) {
    int i = blockIdx.x * 256 + threadIdx.x;
    if (i < n) dst[i] = f32_to_bf16(src[i]);
}

__global__ __launch_bounds__(256) void k_add_bias(const float* __restrict__ a,
                                                  const float* __restrict__ b,
                                                  float* __restrict__ dst, int n) {
    int i = blockIdx.x * 256 + threadIdx.x;
    if (i < n) dst[i] = a[i] + b[i];
}

__global__ __launch_bounds__(256) void k_zero_u16(unsigned short* __restrict__ p, int n) {
    int i = blockIdx.x * 256 + threadIdx.x;
    if (i < n) p[i] = 0;
}

// ---------------- encoder layer 0 step ----------------
// h_out = tanh( x[:,t,:] @ Wih0^T + bias0 + h_in @ Whh0^T )   (B x H)
__global__ __launch_bounds__(256)
void enc_l0_step(const unsigned short* __restrict__ h_in,
                 unsigned short* __restrict__ h_out,
                 const unsigned short* __restrict__ Whh0b,
                 const float* __restrict__ x,
                 const float* __restrict__ Wih0,
                 const float* __restrict__ bias0,
                 int t) {
    int wave = threadIdx.x >> 5;
    int lane = threadIdx.x & 31;
    int tid  = blockIdx.x * 8 + wave;        // 0..511 tiles
    int tileN = tid & 31;                    // H/16 = 32
    int tileM = tid >> 5;                    // B/16 = 16
    int m0 = tileM * 16, n0 = tileN * 16;

    v8f acc = {};
    #pragma unroll 4
    for (int k0 = 0; k0 < Hdim; k0 += 32) {
        v16bf a = load_tile(h_in,  m0, k0, lane);
        v16bf b = load_tile(Whh0b, n0, k0, lane);
        acc = __builtin_amdgcn_wmma_f32_16x16x32_bf16(false, a, false, b,
                                                      (short)0, acc, false, false);
    }

    int half = lane >> 4;
    int j = n0 + (lane & 15);
    float bj = bias0[j];
    float w0 = Wih0[j * 4 + 0], w1 = Wih0[j * 4 + 1];
    float w2 = Wih0[j * 4 + 2], w3 = Wih0[j * 4 + 3];
    #pragma unroll
    for (int v = 0; v < 8; ++v) {
        int brow = m0 + v + half * 8;
        const float* xp = x + ((size_t)brow * Sdim + t) * Vdim;
        float val = acc[v] + bj + xp[0] * w0 + xp[1] * w1 + xp[2] * w2 + xp[3] * w3;
        h_out[(size_t)brow * Hdim + j] = f32_to_bf16(tanhf(val));
    }
}

// ---------------- encoder layer 1 step ----------------
// h2_out = tanh( y1 @ Wih1^T + h2_in @ Whh1^T + bias1 );  also fp32 copy -> h2f
__global__ __launch_bounds__(256)
void enc_l1_step(const unsigned short* __restrict__ y1,
                 const unsigned short* __restrict__ h2_in,
                 unsigned short* __restrict__ h2_out,
                 float* __restrict__ h2f,
                 const unsigned short* __restrict__ Wih1b,
                 const unsigned short* __restrict__ Whh1b,
                 const float* __restrict__ bias1) {
    int wave = threadIdx.x >> 5;
    int lane = threadIdx.x & 31;
    int tid  = blockIdx.x * 8 + wave;
    int tileN = tid & 31;
    int tileM = tid >> 5;
    int m0 = tileM * 16, n0 = tileN * 16;

    v8f acc = {};
    #pragma unroll 2
    for (int k0 = 0; k0 < Hdim; k0 += 32) {
        v16bf a0 = load_tile(y1,    m0, k0, lane);
        v16bf b0 = load_tile(Wih1b, n0, k0, lane);
        acc = __builtin_amdgcn_wmma_f32_16x16x32_bf16(false, a0, false, b0,
                                                      (short)0, acc, false, false);
        v16bf a1 = load_tile(h2_in, m0, k0, lane);
        v16bf b1 = load_tile(Whh1b, n0, k0, lane);
        acc = __builtin_amdgcn_wmma_f32_16x16x32_bf16(false, a1, false, b1,
                                                      (short)0, acc, false, false);
    }

    int half = lane >> 4;
    int j = n0 + (lane & 15);
    float bj = bias1[j];
    #pragma unroll
    for (int v = 0; v < 8; ++v) {
        int brow = m0 + v + half * 8;
        float th = tanhf(acc[v] + bj);
        h2_out[(size_t)brow * Hdim + j] = f32_to_bf16(th);
        h2f[(size_t)brow * Hdim + j] = th;
    }
}

// ---------------- arctanh + softmax over batch axis (per H column) ----------------
__global__ __launch_bounds__(256)
void k_softmax_batch(const float* __restrict__ h2f, float* __restrict__ sm) {
    int j = blockIdx.x;      // H columns
    int b = threadIdx.x;     // B = 256
    float v = atanhf(h2f[(size_t)b * Hdim + j]);
    __shared__ float red[256];
    red[b] = v;
    __syncthreads();
    for (int s = 128; s > 0; s >>= 1) {
        if (b < s) red[b] = fmaxf(red[b], red[b + s]);
        __syncthreads();
    }
    float mx = red[0];
    __syncthreads();
    float e = __expf(v - mx);
    red[b] = e;
    __syncthreads();
    for (int s = 128; s > 0; s >>= 1) {
        if (b < s) red[b] += red[b + s];
        __syncthreads();
    }
    sm[(size_t)b * Hdim + j] = e / red[0];
}

// ---------------- a1 = sm @ d_Wih0^T + d_bih0   (B x 4) ----------------
__global__ __launch_bounds__(256)
void k_dec_a1(const float* __restrict__ sm, const float* __restrict__ dWih0,
              const float* __restrict__ dbih0, float* __restrict__ a1) {
    int b = threadIdx.x;
    const float* row = sm + (size_t)b * Hdim;
    float s0 = 0.f, s1 = 0.f, s2 = 0.f, s3 = 0.f;
    for (int h = 0; h < Hdim; ++h) {
        float sv = row[h];
        s0 += sv * dWih0[0 * Hdim + h];
        s1 += sv * dWih0[1 * Hdim + h];
        s2 += sv * dWih0[2 * Hdim + h];
        s3 += sv * dWih0[3 * Hdim + h];
    }
    a1[b * 4 + 0] = s0 + dbih0[0];
    a1[b * 4 + 1] = s1 + dbih0[1];
    a1[b * 4 + 2] = s2 + dbih0[2];
    a1[b * 4 + 3] = s3 + dbih0[3];
}

// ---------------- decoder: 512 steps, one batch row per thread ----------------
__global__ __launch_bounds__(256)
void k_decoder(const float* __restrict__ a1,
               const float* __restrict__ dWhh0, const float* __restrict__ dbhh0,
               const float* __restrict__ dWih1, const float* __restrict__ dbih1,
               const float* __restrict__ dWhh1, const float* __restrict__ dbhh1,
               float* __restrict__ out) {
    int b = threadIdx.x;
    float W0[16], W1[16], W2[16], b0[4], b1[4], A[4];
    #pragma unroll
    for (int i = 0; i < 16; ++i) { W0[i] = dWhh0[i]; W1[i] = dWih1[i]; W2[i] = dWhh1[i]; }
    #pragma unroll
    for (int v = 0; v < 4; ++v) { b0[v] = dbhh0[v]; b1[v] = dbih1[v] + dbhh1[v]; A[v] = a1[b * 4 + v]; }
    float h1[4] = {0.f, 0.f, 0.f, 0.f}, h2[4] = {0.f, 0.f, 0.f, 0.f};
    for (int t = 0; t < Sdim; ++t) {
        float n1[4], n2[4];
        #pragma unroll
        for (int v = 0; v < 4; ++v)
            n1[v] = tanhf(A[v] + b0[v] + W0[v*4+0]*h1[0] + W0[v*4+1]*h1[1]
                                        + W0[v*4+2]*h1[2] + W0[v*4+3]*h1[3]);
        #pragma unroll
        for (int v = 0; v < 4; ++v)
            n2[v] = tanhf(b1[v] + W1[v*4+0]*n1[0] + W1[v*4+1]*n1[1]
                                 + W1[v*4+2]*n1[2] + W1[v*4+3]*n1[3]
                                 + W2[v*4+0]*h2[0] + W2[v*4+1]*h2[1]
                                 + W2[v*4+2]*h2[2] + W2[v*4+3]*h2[3]);
        #pragma unroll
        for (int v = 0; v < 4; ++v) { h1[v] = n1[v]; h2[v] = n2[v]; }
        float4* op = (float4*)(out + ((size_t)b * Sdim + t) * Vdim);
        *op = make_float4(n2[0], n2[1], n2[2], n2[3]);
    }
}

extern "C" void kernel_launch(void* const* d_in, const int* in_sizes, int n_in,
                              void* d_out, int out_size, void* d_ws, size_t ws_size,
                              hipStream_t stream) {
    const float* x      = (const float*)d_in[0];
    const float* eWih0  = (const float*)d_in[1];
    const float* eWhh0  = (const float*)d_in[2];
    const float* ebih0  = (const float*)d_in[3];
    const float* ebhh0  = (const float*)d_in[4];
    const float* eWih1  = (const float*)d_in[5];
    const float* eWhh1  = (const float*)d_in[6];
    const float* ebih1  = (const float*)d_in[7];
    const float* ebhh1  = (const float*)d_in[8];
    const float* dWih0  = (const float*)d_in[9];
    const float* dWhh0  = (const float*)d_in[10];
    const float* dbih0  = (const float*)d_in[11];
    const float* dbhh0  = (const float*)d_in[12];
    const float* dWih1  = (const float*)d_in[13];
    const float* dWhh1  = (const float*)d_in[14];
    const float* dbih1  = (const float*)d_in[15];
    const float* dbhh1  = (const float*)d_in[16];

    // workspace layout (all counts even -> alignment preserved; base is aligned)
    unsigned short* Whh0b = (unsigned short*)d_ws;
    unsigned short* Wih1b = Whh0b + Hdim * Hdim;
    unsigned short* Whh1b = Wih1b + Hdim * Hdim;
    unsigned short* h1p   = Whh1b + Hdim * Hdim;     // ping
    unsigned short* h1q   = h1p + Bdim * Hdim;       // pong
    unsigned short* h2p   = h1q + Bdim * Hdim;
    unsigned short* h2q   = h2p + Bdim * Hdim;
    float* h2f   = (float*)(h2q + Bdim * Hdim);
    float* smbuf = h2f + Bdim * Hdim;
    float* a1buf = smbuf + Bdim * Hdim;
    float* bias0 = a1buf + Bdim * Vdim;
    float* bias1 = bias0 + Hdim;

    const int NW = Hdim * Hdim;                      // 262144
    k_f32_to_bf16<<<(NW + 255) / 256, 256, 0, stream>>>(eWhh0, Whh0b, NW);
    k_f32_to_bf16<<<(NW + 255) / 256, 256, 0, stream>>>(eWih1, Wih1b, NW);
    k_f32_to_bf16<<<(NW + 255) / 256, 256, 0, stream>>>(eWhh1, Whh1b, NW);
    k_add_bias<<<2, 256, 0, stream>>>(ebih0, ebhh0, bias0, Hdim);
    k_add_bias<<<2, 256, 0, stream>>>(ebih1, ebhh1, bias1, Hdim);
    k_zero_u16<<<(Bdim * Hdim + 255) / 256, 256, 0, stream>>>(h1p, Bdim * Hdim);
    k_zero_u16<<<(Bdim * Hdim + 255) / 256, 256, 0, stream>>>(h2p, Bdim * Hdim);

    unsigned short *h1a = h1p, *h1b = h1q, *h2a = h2p, *h2b = h2q;
    for (int t = 0; t < Sdim; ++t) {
        enc_l0_step<<<64, 256, 0, stream>>>(h1a, h1b, Whh0b, x, eWih0, bias0, t);
        enc_l1_step<<<64, 256, 0, stream>>>(h1b, h2a, h2b, h2f, Wih1b, Whh1b, bias1);
        unsigned short* tmp;
        tmp = h1a; h1a = h1b; h1b = tmp;
        tmp = h2a; h2a = h2b; h2b = tmp;
    }

    k_softmax_batch<<<Hdim, 256, 0, stream>>>(h2f, smbuf);
    k_dec_a1<<<1, 256, 0, stream>>>(smbuf, dWih0, dbih0, a1buf);
    k_decoder<<<1, 256, 0, stream>>>(a1buf, dWhh0, dbhh0, dWih1, dbih1,
                                     dWhh1, dbhh1, (float*)d_out);
}
